// LocalAttentionND_77043123355932
// MI455X (gfx1250) — matrix operand order repaired and checked by reading, not compiled, not run on hardware
//
#include <hip/hip_runtime.h>
#include <math.h>

// Problem constants from the reference
#define BB 2
#define LL 4096
#define CC 512
#define HH 8
#define DD 64
#define KWIN 49
#define NROWS (BB * LL)          // 8192
#define SCALE_QK 0.125f          // D^-0.5, D=64
#define LTILES (LL / 16)         // 256

typedef __attribute__((ext_vector_type(16))) __bf16 bf16x16;
typedef __attribute__((ext_vector_type(8)))  __bf16 bf16x8;
typedef __attribute__((ext_vector_type(8)))  float  f32x8;
typedef unsigned short u16;

// ---------- device helpers ----------
__device__ __forceinline__ u16 f2bf(float f) {
  union { float f; unsigned u; } a; a.f = f;
  unsigned u = a.u;
  u += 0x7FFFu + ((u >> 16) & 1u);   // round-to-nearest-even
  return (u16)(u >> 16);
}
__device__ __forceinline__ float sigm(float x) { return 1.0f / (1.0f + expf(-x)); }
__device__ __forceinline__ float silu(float x) { return x / (1.0f + expf(-x)); }

__device__ __forceinline__ f32x8 wmma_bf16(bf16x16 a, bf16x16 b, f32x8 c) {
  // D = A(16x32) * B(32x16) + C, f32 accumulate
  return __builtin_amdgcn_wmma_f32_16x16x32_bf16(false, a, false, b, (short)0, c,
                                                 false, false);
}

// Async global -> LDS bulk copy (gfx1250, ASYNCcnt-tracked).
// lds_off is the raw LDS byte address (dynamic-LDS region starts at 0 when the
// kernel has no static __shared__), gaddr is the 64-bit global byte address.
__device__ __forceinline__ void async_g2l_b128(unsigned lds_off, unsigned long long gaddr) {
  asm volatile("global_load_async_to_lds_b128 %0, %1, off"
               :: "v"(lds_off), "v"(gaddr)
               : "memory");
}
__device__ __forceinline__ void wait_asynccnt0() {
  asm volatile("s_wait_asynccnt 0x0" ::: "memory");
}

// ---------- fp32 -> bf16 converters ----------
__global__ void cvt_f32_bf16(const float* __restrict__ src, u16* __restrict__ dst, int n) {
  int i = blockIdx.x * blockDim.x + threadIdx.x;
  if (i < n) dst[i] = f2bf(src[i]);
}

// extract V half of kv (row stride 1024, offset 512) into contiguous bf16
__global__ void cvt_v_bf16(const float* __restrict__ kv, u16* __restrict__ vbf) {
  int i = blockIdx.x * blockDim.x + threadIdx.x;
  if (i < NROWS * CC) {
    int row = i >> 9, c = i & 511;
    vbf[i] = f2bf(kv[(size_t)row * (2 * CC) + CC + c]);
  }
}

// ---------- generic bf16 WMMA GEMM: Y = act(X * W^T + bias) ----------
// X: (rows, Kdim) bf16, W: (Mcols, Kdim) bf16, Y: (rows, Mcols) f32
// One wave computes a 32x64 output tile: 2 row-tiles share every B-tile load
// (8 wmma per 32-K step, ~21 FLOP per L2 byte).
__global__ void __launch_bounds__(128)
gemm_bf16_silu(const u16* __restrict__ X, const u16* __restrict__ W,
               const float* __restrict__ bias, float* __restrict__ Y,
               int Mcols, int Kdim, int act) {
  int wid  = (blockIdx.x * blockDim.x + threadIdx.x) >> 5;
  int lane = threadIdx.x & 31;
  int strips = Mcols >> 6;
  int rt = wid / strips;
  int cs = wid % strips;
  int row0 = rt << 5;              // 32 rows per wave
  int col0 = cs << 6;              // 64 cols per wave
  int mrow = lane & 15;
  int hi   = lane >> 4;

  f32x8 acc[8];                    // [t] rows 0-15, [4+t] rows 16-31
#pragma unroll
  for (int t = 0; t < 8; ++t) acc[t] = (f32x8){0.f,0.f,0.f,0.f,0.f,0.f,0.f,0.f};

  const u16* xrow0 = X + (size_t)(row0 + mrow) * Kdim;
  const u16* xrow1 = X + (size_t)(row0 + 16 + mrow) * Kdim;
  for (int k0 = 0; k0 < Kdim; k0 += 32) {
    if (k0 + 64 < Kdim) {
      __builtin_prefetch(xrow0 + k0 + 64, 0, 3);   // near-cache prefetch
      __builtin_prefetch(xrow1 + k0 + 64, 0, 3);
    }
    // A tiles (16 rows x 32 K each): lane holds two contiguous 8-elt chunks
    union { bf16x16 v; bf16x8 h[2]; } A0, A1;
    A0.h[0] = *reinterpret_cast<const bf16x8*>(xrow0 + k0 + hi * 16);
    A0.h[1] = *reinterpret_cast<const bf16x8*>(xrow0 + k0 + hi * 16 + 8);
    A1.h[0] = *reinterpret_cast<const bf16x8*>(xrow1 + k0 + hi * 16);
    A1.h[1] = *reinterpret_cast<const bf16x8*>(xrow1 + k0 + hi * 16 + 8);
#pragma unroll
    for (int t = 0; t < 4; ++t) {
      // B tile (32 K x 16 N): lane = out col, 16 contiguous K elems at hi*16
      const u16* wrow = W + (size_t)(col0 + t * 16 + mrow) * Kdim + k0 + hi * 16;
      bf16x16 Bv = *reinterpret_cast<const bf16x16*>(wrow);
      acc[t]     = wmma_bf16(A0.v, Bv, acc[t]);
      acc[4 + t] = wmma_bf16(A1.v, Bv, acc[4 + t]);
    }
  }

#pragma unroll
  for (int t = 0; t < 4; ++t) {
    int col = col0 + t * 16 + mrow;
    float bs = bias ? bias[col] : 0.0f;
#pragma unroll
    for (int r = 0; r < 8; ++r) {
      int rowA = row0 + r + 8 * hi;
      float v0 = acc[t][r] + bs;
      if (act) v0 = silu(v0);
      Y[(size_t)rowA * Mcols + col] = v0;
      float v1 = acc[4 + t][r] + bs;
      if (act) v1 = silu(v1);
      Y[(size_t)(rowA + 16) * Mcols + col] = v1;
    }
  }
}

// ---------- window params: wp = silu(x @ Wwin^T + bwin) -> width, sharp ----------
__global__ void winparams_kernel(const float* __restrict__ x, const float* __restrict__ Wwin,
                                 const float* __restrict__ bwin,
                                 float* __restrict__ width, float* __restrict__ sharp) {
  int idx = blockIdx.x * blockDim.x + threadIdx.x;
  if (idx >= NROWS * 2 * HH) return;
  int row = idx >> 4;
  int j   = idx & 15;
  const float* xr = x + (size_t)row * CC;
  const float* wr = Wwin + (size_t)j * CC;
  float d = 0.f;
  for (int c = 0; c < CC; ++c) d += xr[c] * wr[c];
  float s = silu(d + bwin[j]);
  if (j < HH) width[(size_t)row * HH + j] = sigm(s) * 24.0f + 0.5f;
  else        sharp[(size_t)row * HH + (j - HH)] = sigm(s) * 9.5f + 0.5f;
}

// ---------- per-head RMSNorm + RoPE -> bf16 ----------
__global__ void qknorm_rope_kernel(const float* __restrict__ src, int rowStride,
                                   const float* __restrict__ nw, u16* __restrict__ dst) {
  int idx = blockIdx.x * blockDim.x + threadIdx.x;
  if (idx >= NROWS * HH) return;
  int row = idx >> 3;
  int h   = idx & 7;
  const float* base = src + (size_t)row * rowStride + h * DD;
  float ssq = 0.f;
  for (int i = 0; i < DD; ++i) { float v = base[i]; ssq += v * v; }
  float inv = 1.0f / sqrtf(ssq * (1.0f / DD) + 1e-6f);
  int l = row & (LL - 1);
  u16* drow = dst + (size_t)row * CC + h * DD;
  for (int i = 0; i < 32; ++i) {
    float x1 = base[i]      * inv * nw[i];
    float x2 = base[i + 32] * inv * nw[i + 32];
    float freq = expf(-(float)i * (9.210340371976184f / 32.0f)); // 10000^(-i/32)
    float ang = (float)l * freq;
    float c = cosf(ang), s = sinf(ang);
    drow[i]      = f2bf(x1 * c - x2 * s);
    drow[i + 32] = f2bf(x1 * s + x2 * c);
  }
}

// ---------- windowed attention via WMMA + async V staging ----------
// One wave per (b, 16-row l-tile, head); 49-wide windows of 16 rows span 64 keys.
// Dynamic LDS layout (base offset 0, no static __shared__ in this kernel):
//   [0     .. 8192)  Vs: 64 key-rows x 64 bf16 (row stride 128 B)
//   [8192  .. 12288) Sb: 16 x 64 f32 raw scores
//   [12288 .. 14336) Ab: 16 x 64 bf16 attention weights
#define ATTN_LDS_BYTES 14336
__global__ void __launch_bounds__(32)
attn_kernel(const u16* __restrict__ qbf, const u16* __restrict__ kbf,
            const u16* __restrict__ vbf, const float* __restrict__ width,
            const float* __restrict__ sharp, float* __restrict__ outp) {
  extern __shared__ __align__(16) unsigned char smem[];
  u16*   Vs = (u16*)(smem);
  float* Sb = (float*)(smem + 8192);
  u16*   Ab = (u16*)(smem + 12288);

  int lane = threadIdx.x & 31;
  int wid  = blockIdx.x;
  int h  = wid & 7;
  int lt = (wid >> 3) & (LTILES - 1);
  int b  = wid >> 11;
  int l0 = lt << 4;
  int mrow = lane & 15;
  int hi   = lane >> 4;
  size_t baseRow = (size_t)b * LL;

  // ---- kick off async staging of the 64x64 V span into LDS (row-major) ----
  // Per step: 32 lanes x 16 B = 4 key-rows; 16 steps cover 64 rows (8 KB).
  {
    int rlane = lane >> 3;          // 0..3 : row within group of 4
    int coff  = (lane & 7) * 16;    // byte offset within 128-B row
#pragma unroll
    for (int i = 0; i < 16; ++i) {
      int r = i * 4 + rlane;
      int p = l0 - 24 + r;
      unsigned off = (unsigned)(r * 128 + coff);
      if (p >= 0 && p < LL) {
        unsigned long long g = (unsigned long long)(size_t)
            ((const unsigned char*)(vbf + (baseRow + p) * CC + h * DD) + coff);
        async_g2l_b128(off, g);
      } else {
        *(uint4*)(smem + off) = make_uint4(0u, 0u, 0u, 0u);  // jnp.pad zeros
      }
    }
  }

  // ---- scores: S = Q(16x64) * Kspan^T(64x64) via 2 d-chunks x 4 col tiles ----
  f32x8 sacc[4];
#pragma unroll
  for (int t = 0; t < 4; ++t) sacc[t] = (f32x8){0.f,0.f,0.f,0.f,0.f,0.f,0.f,0.f};
  const u16* qrow = qbf + (baseRow + l0 + mrow) * CC + h * DD;
#pragma unroll
  for (int dc = 0; dc < 2; ++dc) {
    union { bf16x16 v; bf16x8 hh[2]; } A;
    A.hh[0] = *reinterpret_cast<const bf16x8*>(qrow + dc * 32 + hi * 16);
    A.hh[1] = *reinterpret_cast<const bf16x8*>(qrow + dc * 32 + hi * 16 + 8);
#pragma unroll
    for (int t = 0; t < 4; ++t) {
      int j = t * 16 + mrow;           // key index in 64-span (B matrix col)
      int p = l0 - 24 + j;             // global key position
      union { bf16x16 v; unsigned u[8]; } Bv;
      if (p >= 0 && p < LL) {
        Bv.v = *reinterpret_cast<const bf16x16*>(
            kbf + (baseRow + p) * CC + h * DD + dc * 32 + hi * 16);
      } else {
#pragma unroll
        for (int i = 0; i < 8; ++i) Bv.u[i] = 0u;   // zero padding (matches jnp.pad)
      }
      sacc[t] = wmma_bf16(A.v, Bv.v, sacc[t]);
    }
  }
#pragma unroll
  for (int t = 0; t < 4; ++t)
#pragma unroll
    for (int r = 0; r < 8; ++r)
      Sb[(r + 8 * hi) * 64 + t * 16 + mrow] = sacc[t][r] * SCALE_QK;
  __syncthreads();

  // ---- softmax with soft distance mask (lanes 0..15, one row each) ----
  if (lane < 16) {
    int m = lane;
    int grow = l0 + m;
    float wd = width[(baseRow + grow) * HH + h];
    float sp = sharp[(baseRow + grow) * HH + h];
    float msc[KWIN];
    float mx = -1e30f;
#pragma unroll
    for (int w = 0; w < KWIN; ++w) {
      float s = Sb[m * 64 + m + w];
      float rel = fabsf((float)w - 24.0f);
      float sm = sigm((wd - rel) * sp);
      s -= (1.0f - sm) * 10000.0f;
      msc[w] = s;
      mx = fmaxf(mx, s);
    }
    float sum = 0.f;
#pragma unroll
    for (int w = 0; w < KWIN; ++w) { float e = expf(msc[w] - mx); msc[w] = e; sum += e; }
    float inv = 1.0f / sum;
    for (int j = 0; j < 64; ++j) Ab[m * 64 + j] = 0;   // outside band
#pragma unroll
    for (int w = 0; w < KWIN; ++w) Ab[m * 64 + m + w] = f2bf(msc[w] * inv);
  }
  wait_asynccnt0();   // V span resident in LDS
  __syncthreads();

  // ---- out = Attn(16x64) * Vspan(64x64), V gathered from LDS ----
  f32x8 oacc[4];
#pragma unroll
  for (int t = 0; t < 4; ++t) oacc[t] = (f32x8){0.f,0.f,0.f,0.f,0.f,0.f,0.f,0.f};
#pragma unroll
  for (int jc = 0; jc < 2; ++jc) {
    int j0 = jc * 32;
    union { bf16x16 v; bf16x8 hh[2]; } A;
    A.hh[0] = *reinterpret_cast<const bf16x8*>(Ab + mrow * 64 + j0 + hi * 8);
    A.hh[1] = *reinterpret_cast<const bf16x8*>(Ab + mrow * 64 + j0 + 16 + hi * 8);
#pragma unroll
    for (int t = 0; t < 4; ++t) {
      int d = t * 16 + mrow;
      union { bf16x16 v; u16 u[16]; } Bv;
#pragma unroll
      for (int e = 0; e < 16; ++e) {
        int j = j0 + (e < 8 ? hi * 8 + e : 16 + hi * 8 + (e - 8));
        Bv.u[e] = Vs[j * 64 + d];      // ds_load_u16 gather (transpose in LDS)
      }
      oacc[t] = wmma_bf16(A.v, Bv.v, oacc[t]);
    }
  }
#pragma unroll
  for (int t = 0; t < 4; ++t)
#pragma unroll
    for (int r = 0; r < 8; ++r) {
      int row = l0 + r + 8 * hi;
      outp[(baseRow + row) * CC + h * DD + t * 16 + mrow] = oacc[t][r];
    }
}

// ---------- RMSNorm(out) + gate merge -> merged bf16 ----------
__global__ void __launch_bounds__(256)
merge_kernel(const float* __restrict__ attn, const float* __restrict__ gpre,
             const float* __restrict__ kv, const float* __restrict__ onw,
             u16* __restrict__ mbf) {
  __shared__ float red[256];
  int row = blockIdx.x;
  int tid = threadIdx.x;
  float a0 = attn[(size_t)row * CC + tid];
  float a1 = attn[(size_t)row * CC + tid + 256];
  red[tid] = a0 * a0 + a1 * a1;
  __syncthreads();
  for (int s = 128; s > 0; s >>= 1) {
    if (tid < s) red[tid] += red[tid + s];
    __syncthreads();
  }
  float inv = 1.0f / sqrtf(red[0] * (1.0f / CC) + 1e-6f);
#pragma unroll
  for (int q = 0; q < 2; ++q) {
    int c = tid + q * 256;
    float a = q ? a1 : a0;
    float on = a * inv * onw[c];
    float g = sigm(gpre[(size_t)row * CC + c]);
    float v = kv[(size_t)row * (2 * CC) + CC + c];
    mbf[(size_t)row * CC + c] = f2bf(g * v + (1.0f - g) * on);
  }
}

// ---------- host launch ----------
extern "C" void kernel_launch(void* const* d_in, const int* in_sizes, int n_in,
                              void* d_out, int out_size, void* d_ws, size_t ws_size,
                              hipStream_t stream) {
  const float* x    = (const float*)d_in[0];
  const float* Wq   = (const float*)d_in[1];
  const float* Wkv  = (const float*)d_in[2];
  const float* Wwin = (const float*)d_in[3];
  const float* bwin = (const float*)d_in[4];
  const float* Wg   = (const float*)d_in[5];
  const float* bg   = (const float*)d_in[6];
  const float* Wout = (const float*)d_in[7];
  const float* qn   = (const float*)d_in[8];
  const float* kn   = (const float*)d_in[9];
  const float* on   = (const float*)d_in[10];
  float* out = (float*)d_out;

  char* p = (char*)d_ws;
  auto alloc = [&](size_t bytes) -> char* {
    char* r = p; p += (bytes + 255) & ~(size_t)255; return r;
  };
  u16*   xbf   = (u16*)  alloc((size_t)NROWS * CC * 2);
  u16*   wqbf  = (u16*)  alloc((size_t)CC * CC * 2);
  u16*   wkvbf = (u16*)  alloc((size_t)2 * CC * CC * 2);
  u16*   wgbf  = (u16*)  alloc((size_t)CC * CC * 2);
  u16*   wobf  = (u16*)  alloc((size_t)CC * CC * 2);
  float* qf    = (float*)alloc((size_t)NROWS * CC * 4);
  float* kvf   = (float*)alloc((size_t)NROWS * 2 * CC * 4);
  u16*   qbf   = (u16*)  alloc((size_t)NROWS * CC * 2);
  u16*   kbf   = (u16*)  alloc((size_t)NROWS * CC * 2);
  u16*   vbf   = (u16*)  alloc((size_t)NROWS * CC * 2);
  float* widthb= (float*)alloc((size_t)NROWS * HH * 4);
  float* sharpb= (float*)alloc((size_t)NROWS * HH * 4);
  float* attnf = (float*)alloc((size_t)NROWS * CC * 4);
  float* gpre  = (float*)alloc((size_t)NROWS * CC * 4);
  u16*   mbf   = (u16*)  alloc((size_t)NROWS * CC * 2);
  (void)ws_size; (void)in_sizes; (void)n_in; (void)out_size;

  const int T = 256;
  cvt_f32_bf16<<<(NROWS * CC + T - 1) / T, T, 0, stream>>>(x, xbf, NROWS * CC);
  cvt_f32_bf16<<<(CC * CC + T - 1) / T, T, 0, stream>>>(Wq, wqbf, CC * CC);
  cvt_f32_bf16<<<(2 * CC * CC + T - 1) / T, T, 0, stream>>>(Wkv, wkvbf, 2 * CC * CC);
  cvt_f32_bf16<<<(CC * CC + T - 1) / T, T, 0, stream>>>(Wg, wgbf, CC * CC);
  cvt_f32_bf16<<<(CC * CC + T - 1) / T, T, 0, stream>>>(Wout, wobf, CC * CC);

  // q = silu(x @ Wq^T)   : 32x64 tiles -> (8192/32)*(512/64) = 2048 waves
  gemm_bf16_silu<<<(NROWS / 32) * (CC / 64) / 4, 128, 0, stream>>>(
      xbf, wqbf, nullptr, qf, CC, CC, 1);
  // kv = silu(x @ Wkv^T) : 4096 waves
  gemm_bf16_silu<<<(NROWS / 32) * (2 * CC / 64) / 4, 128, 0, stream>>>(
      xbf, wkvbf, nullptr, kvf, 2 * CC, CC, 1);

  winparams_kernel<<<(NROWS * 2 * HH + T - 1) / T, T, 0, stream>>>(
      x, Wwin, bwin, widthb, sharpb);

  qknorm_rope_kernel<<<(NROWS * HH + T - 1) / T, T, 0, stream>>>(qf, CC, qn, qbf);
  qknorm_rope_kernel<<<(NROWS * HH + T - 1) / T, T, 0, stream>>>(kvf, 2 * CC, kn, kbf);
  cvt_v_bf16<<<(NROWS * CC + T - 1) / T, T, 0, stream>>>(kvf, vbf);

  // attention: one wave per (b, l-tile, head) = 4096 workgroups of 32 threads
  attn_kernel<<<BB * LTILES * HH, 32, ATTN_LDS_BYTES, stream>>>(
      qbf, kbf, vbf, widthb, sharpb, attnf);

  // gate_pre = silu(v @ Wg^T + bg)
  gemm_bf16_silu<<<(NROWS / 32) * (CC / 64) / 4, 128, 0, stream>>>(
      vbf, wgbf, bg, gpre, CC, CC, 1);

  merge_kernel<<<NROWS, 256, 0, stream>>>(attnf, gpre, kvf, on, mbf);

  // final: silu(merged @ Wout^T)
  gemm_bf16_silu<<<(NROWS / 32) * (CC / 64) / 4, 128, 0, stream>>>(
      mbf, wobf, nullptr, out, CC, CC, 1);
}
